// RelativeAttention_3152505995806
// MI455X (gfx1250) — compile-verified, matrix operand order checked by reading, not run on hardware
//
#include <hip/hip_runtime.h>
#include <hip/hip_bf16.h>

// Problem constants
#define BB   4
#define TT   1024
#define NXX  1024
#define HH   16
#define DD   64
#define QREL_N 65   // 2*KMAX+1

typedef __attribute__((ext_vector_type(16))) __bf16 bf16x16;
typedef __attribute__((ext_vector_type(8)))  float  f32x8;
typedef __attribute__((ext_vector_type(4)))  unsigned u32x4;
typedef __attribute__((ext_vector_type(8)))  unsigned u32x8;

union AFrag {
  bf16x16 v;
  unsigned u[8];
  unsigned short h[16];
};

__device__ __forceinline__ unsigned short f2bf_u(float f) {
  union { float f; unsigned u; } x; x.f = f;
  unsigned r = x.u + 0x7FFFu + ((x.u >> 16) & 1u);
  return (unsigned short)(r >> 16);
}
__device__ __forceinline__ float bf2f_u(unsigned short s) {
  union { float f; unsigned u; } x; x.u = ((unsigned)s) << 16; return x.f;
}

__device__ __forceinline__ f32x8 wmma_bf16(const AFrag& a, const AFrag& b, f32x8 c) {
  return __builtin_amdgcn_wmma_f32_16x16x32_bf16(false, a.v, false, b.v, (short)0, c,
                                                 false, false);
}

// ISA 16-bit A/B fragment layout (16x32): lane<16 holds K{0..7,16..23},
// lane>=16 holds K{8..15,24..31}; 2 bf16 per VGPR.
__device__ __forceinline__ int kpair(int p, int kbase) {        // even k of dword p
  return (p < 4) ? (kbase + 2 * p) : (16 + kbase + 2 * (p - 4));
}

// ---- Tensor Data Mover helpers (D# groups per CDNA5 ISA) ----
// 2D bf16 tensor [rows][K] row-major; tile = 32 k-elements (contig) x tile_rows.
__device__ __forceinline__ u32x8 tdm_desc_g1(int rows, int tile_rows, int K) {
  u32x8 g1;
  g1[0] = 0x00010000u;                                              // data_size=1 (2B)
  g1[1] = ((unsigned)K & 0xFFFFu) << 16;                            // tensor_dim0 lo
  g1[2] = (((unsigned)K >> 16) & 0xFFFFu) | (((unsigned)rows & 0xFFFFu) << 16);
  g1[3] = (((unsigned)rows >> 16) & 0xFFFFu) | (32u << 16);         // tile_dim0=32
  g1[4] = (unsigned)tile_rows;                                      // tile_dim1
  g1[5] = (unsigned)K;                                              // dim0 stride lo
  g1[6] = 0u;
  g1[7] = 0u;
  return g1;
}
__device__ __forceinline__ void tdm_load_2d(unsigned lds_off, const void* gaddr,
                                            const u32x8& g1) {
  size_t ga = (size_t)gaddr;
  u32x4 g0;
  g0[0] = 1u;                                       // count=1, user descriptor
  g0[1] = lds_off;                                  // lds_addr
  g0[2] = (unsigned)ga;                             // global_addr[31:0]
  g0[3] = ((unsigned)(ga >> 32) & 0x01FFFFFFu) | (2u << 30);  // addr hi | type=2
  asm volatile("tensor_load_to_lds %0, %1" :: "s"(g0), "s"(g1) : "memory");
}

// ---------------------------------------------------------------------------
__global__ void cvt_f32_bf16(const float* __restrict__ in,
                             unsigned short* __restrict__ out, int n) {
  int i = blockIdx.x * 256 + threadIdx.x;
  if (i < n) out[i] = f2bf_u(in[i]);
}

// Transpose-convert weights: in [K][N] f32 -> out [N][K] bf16 (paid once).
__global__ void cvt_transpose_bf16(const float* __restrict__ in,
                                   unsigned short* __restrict__ out,
                                   int K, int N) {
  int n = blockIdx.x * 256 + threadIdx.x;
  int k = blockIdx.y;
  if (n < N) out[(size_t)n * K + k] = f2bf_u(in[(size_t)k * N + n]);
}

// ---------------------------------------------------------------------------
// 64x128 block-tile WMMA GEMM, double-buffered Tensor-Data-Mover pipeline.
// C = A(bf16 [M][K]) * B^T(bf16 [N][K], pre-transposed) + bias.
// Wave 0 issues TDM descriptors for tile j+1 while tile j is consumed;
// TENSORcnt (in-order per wave) gates buffer readiness.
// mode 0: QKV epilogue (Q/K in [B,H,T,D]; V transposed to [B,H,D,T])
// mode 1: fp32 output (final projection)
__global__ __launch_bounds__(256)
void gemm_bf16(const unsigned short* __restrict__ A,
               const unsigned short* __restrict__ BT,
               const float* __restrict__ bias,
               int M, int N, int K, int mode,
               float* __restrict__ outF,
               unsigned short* __restrict__ Qb,
               unsigned short* __restrict__ Kb,
               unsigned short* __restrict__ Vt) {
  __shared__ unsigned short As[2][64][32];     // [buf][m][k]
  __shared__ unsigned short Bs[2][128][32];    // [buf][n][k]
  int tid = threadIdx.x;
  int w = tid >> 5, lane = tid & 31;
  int wr = w >> 1, wc = w & 1;                 // 4x2 wave grid -> 16x64 per wave
  int m0 = blockIdx.y * 64, n0 = blockIdx.x * 128;
  int kbase = (lane >> 4) * 8;
  f32x8 cc[4] = {};

  u32x8 g1A = tdm_desc_g1(M, 64, K);
  u32x8 g1B = tdm_desc_g1(N, 128, K);
  unsigned ldsA[2] = { (unsigned)(size_t)&As[0][0][0], (unsigned)(size_t)&As[1][0][0] };
  unsigned ldsB[2] = { (unsigned)(size_t)&Bs[0][0][0], (unsigned)(size_t)&Bs[1][0][0] };
  int nsteps = K >> 5;

  if (w == 0) {                                // prologue: fetch tile 0
    tdm_load_2d(ldsA[0], A + (size_t)m0 * K, g1A);
    tdm_load_2d(ldsB[0], BT + (size_t)n0 * K, g1B);
  }
  for (int j = 0; j < nsteps; ++j) {
    int buf = j & 1;
    if (w == 0) {
      if (j + 1 < nsteps) {                    // prefetch tile j+1 into other buffer
        int k1 = (j + 1) << 5;
        tdm_load_2d(ldsA[buf ^ 1], A + (size_t)m0 * K + k1, g1A);
        tdm_load_2d(ldsB[buf ^ 1], BT + (size_t)n0 * K + k1, g1B);
        __builtin_amdgcn_s_wait_tensorcnt(2);  // tile j's pair complete
      } else {
        __builtin_amdgcn_s_wait_tensorcnt(0);
      }
    }
    __syncthreads();                           // tile j visible to all waves

    AFrag a;
    int m = wr * 16 + (lane & 15);
#pragma unroll
    for (int p = 0; p < 8; ++p)
      a.u[p] = *(const unsigned*)(&As[buf][m][kpair(p, kbase)]);
#pragma unroll
    for (int t = 0; t < 4; ++t) {
      AFrag bfrag;
      int nA = wc * 64 + t * 16 + (lane & 15);
#pragma unroll
      for (int p = 0; p < 8; ++p)
        bfrag.u[p] = *(const unsigned*)(&Bs[buf][nA][kpair(p, kbase)]);
      cc[t] = wmma_bf16(a, bfrag, cc[t]);
    }
    __syncthreads();                           // done with buf before j+2 overwrites
  }

  int hi = lane >> 4;
#pragma unroll
  for (int t = 0; t < 4; ++t) {
    int gc = n0 + wc * 64 + t * 16 + (lane & 15);
    float bv = bias[gc];
#pragma unroll
    for (int v = 0; v < 8; ++v) {
      int gr = m0 + wr * 16 + v + 8 * hi;
      float val = cc[t][v] + bv;
      if (mode == 0) {
        int b = gr >> 10, tt = gr & 1023;
        int which = gc >> 10, wi = gc & 1023;
        int h = wi >> 6, d = wi & 63;
        unsigned short q = f2bf_u(val);
        if (which == 0)
          Qb[(((size_t)(b * HH + h)) * TT + tt) * DD + d] = q;
        else if (which == 1)
          Kb[(((size_t)(b * HH + h)) * TT + tt) * DD + d] = q;
        else                                   // V transposed: [B,H,D,T]
          Vt[(((size_t)(b * HH + h)) * DD + d) * TT + tt] = q;
      } else {
        outF[(size_t)gr * N + gc] = val;
      }
    }
  }
}

// ---------------------------------------------------------------------------
// qrel[b,h,t,0..64] = q . rel_k_emb^T ; lr1[b,h,t,n] = (q . LR_K[h]^T)/sqrt(D)
__global__ __launch_bounds__(128)
void aux_q(const unsigned short* __restrict__ Qb,
           const float* __restrict__ rel_k_emb,
           const float* __restrict__ LR_K,
           float* __restrict__ qrel, float* __restrict__ lr1) {
  __shared__ unsigned short BTl[144][64];  // [n][k]: 0..64 rel_k, 80..143 LR_K
  int tid = threadIdx.x, w = tid >> 5, lane = tid & 31;
  int bh = blockIdx.y, h = bh & 15;
  int t0 = blockIdx.x * 64;
  for (int i = tid; i < 144 * 64; i += 128) {
    int n = i >> 6, k = i & 63;
    float v = 0.f;
    if (n < 65)       v = rel_k_emb[n * 64 + k];
    else if (n >= 80) v = LR_K[(h * 64 + (n - 80)) * 64 + k];
    BTl[n][k] = f2bf_u(v);
  }
  __syncthreads();
  int kbase = (lane >> 4) * 8, hi = lane >> 4;
  AFrag a0, a1;
  int t = t0 + w * 16 + (lane & 15);
  const unsigned* qrow = (const unsigned*)(Qb + ((size_t)bh * TT + t) * DD);
#pragma unroll
  for (int p = 0; p < 8; ++p) {
    int kk = kpair(p, kbase);
    a0.u[p] = qrow[kk >> 1];
    a1.u[p] = qrow[(kk + 32) >> 1];
  }
  for (int j = 0; j < 9; ++j) {
    int nb = j * 16;
    AFrag b0, b1;
    int n = nb + (lane & 15);
#pragma unroll
    for (int p = 0; p < 8; ++p) {
      int kk = kpair(p, kbase);
      b0.u[p] = *(const unsigned*)(&BTl[n][kk]);
      b1.u[p] = *(const unsigned*)(&BTl[n][kk + 32]);
    }
    f32x8 c = {};
    c = wmma_bf16(a0, b0, c);
    c = wmma_bf16(a1, b1, c);
#pragma unroll
    for (int v = 0; v < 8; ++v) {
      int tr = t0 + w * 16 + v + 8 * hi;
      if (nb < 80) {
        if (n < 65) qrel[((size_t)bh * TT + tr) * QREL_N + n] = c[v];
      } else {
        lr1[((size_t)bh * TT + tr) * 64 + (n - 80)] = c[v] * 0.125f;
      }
    }
  }
}

// lr2[b,h,n,s] = (LR_Q[h][n] . k[b,h,s]) / sqrt(D)
__global__ __launch_bounds__(128)
void aux_k(const unsigned short* __restrict__ Kb,
           const float* __restrict__ LR_Q, float* __restrict__ lr2) {
  __shared__ unsigned short Aq[64][64];
  int tid = threadIdx.x, w = tid >> 5, lane = tid & 31;
  int bh = blockIdx.y, h = bh & 15;
  int s0 = blockIdx.x * 64;
  for (int i = tid; i < 64 * 64; i += 128) {
    int n = i >> 6, k = i & 63;
    Aq[n][k] = f2bf_u(LR_Q[(h * 64 + n) * 64 + k]);
  }
  __syncthreads();
  int kbase = (lane >> 4) * 8, hi = lane >> 4;
  AFrag a0, a1;
  int n = w * 16 + (lane & 15);
#pragma unroll
  for (int p = 0; p < 8; ++p) {
    int kk = kpair(p, kbase);
    a0.u[p] = *(const unsigned*)(&Aq[n][kk]);
    a1.u[p] = *(const unsigned*)(&Aq[n][kk + 32]);
  }
  for (int j = 0; j < 4; ++j) {
    int sb = s0 + j * 16;
    AFrag b0, b1;
    int s = sb + (lane & 15);
    const unsigned* krow = (const unsigned*)(Kb + ((size_t)bh * TT + s) * DD);
#pragma unroll
    for (int p = 0; p < 8; ++p) {
      int kk = kpair(p, kbase);
      b0.u[p] = krow[kk >> 1];
      b1.u[p] = krow[(kk + 32) >> 1];
    }
    f32x8 c = {};
    c = wmma_bf16(a0, b0, c);
    c = wmma_bf16(a1, b1, c);
#pragma unroll
    for (int v = 0; v < 8; ++v) {
      int nr = w * 16 + v + 8 * hi;
      lr2[((size_t)bh * 64 + nr) * TT + s] = c[v] * 0.125f;
    }
  }
}

// ---------------------------------------------------------------------------
// Attention: 16 query rows per block, full score strip in LDS (bf16).
__global__ __launch_bounds__(128)
void attn_kernel(const unsigned short* __restrict__ Qb,
                 const unsigned short* __restrict__ Kb,
                 const unsigned short* __restrict__ Vt,   // [B,H,D,T]
                 const int* __restrict__ rel,
                 const float* __restrict__ tds,
                 const int* __restrict__ LR_map,
                 const float* __restrict__ rel_w_emb,
                 const float* __restrict__ rel_v_emb,
                 const float* __restrict__ qrel,
                 const float* __restrict__ lr1,
                 const float* __restrict__ lr2,
                 unsigned short* __restrict__ Ob) {
  __shared__ unsigned short sc[16][1032];  // score / prob strip (padded)
  __shared__ float red[16][8];
  __shared__ float rowmax[16];
  __shared__ float rowsum[16];
  __shared__ float bucket[16][34];         // rel_v distance buckets 0..32
  __shared__ float rvs[16][64];

  int tid = threadIdx.x, w = tid >> 5, lane = tid & 31;
  int bh = blockIdx.y, b = bh >> 4, h = bh & 15;
  int t0 = blockIdx.x * 16;
  int L  = t0 + 16;
  int Lr = (L + 31) & ~31;                 // round to 32 for the PV loop
  for (int i = tid; i < 16 * 34; i += 128) ((float*)bucket)[i] = 0.f;

  int kbase = (lane >> 4) * 8, hi = lane >> 4;
  AFrag a0, a1;
  {
    int t = t0 + (lane & 15);
    const unsigned* qrow = (const unsigned*)(Qb + ((size_t)bh * TT + t) * DD);
#pragma unroll
    for (int p = 0; p < 8; ++p) {
      int kk = kpair(p, kbase);
      a0.u[p] = qrow[kk >> 1];
      a1.u[p] = qrow[(kk + 32) >> 1];
    }
  }

  // ---- phase 1: scores + fused bias epilogue -> LDS (bf16)
  for (int s0 = w * 16; s0 < Lr; s0 += 64) {
    AFrag b0, b1;
    int sl = s0 + (lane & 15);
    const unsigned* krow = (const unsigned*)(Kb + ((size_t)bh * TT + sl) * DD);
#pragma unroll
    for (int p = 0; p < 8; ++p) {
      int kk = kpair(p, kbase);
      b0.u[p] = krow[kk >> 1];
      b1.u[p] = krow[(kk + 32) >> 1];
    }
    f32x8 c = {};
    c = wmma_bf16(a0, b0, c);
    c = wmma_bf16(a1, b1, c);
#pragma unroll
    for (int v = 0; v < 8; ++v) {
      int m = v + 8 * hi;
      int t = t0 + m;
      int s = s0 + (lane & 15);
      float wv;
      if (s > t) {
        wv = -1.0e10f;
      } else {
        int dd = t - s;
        int rid = 32 - (dd < 32 ? dd : 32);
        wv = (c[v] + qrel[((size_t)bh * TT + t) * QREL_N + rid]) * 0.125f;
        wv *= rel_w_emb[rel[((size_t)b * TT + t) * TT + s] * HH + h];
        wv += tds[((size_t)bh * TT + t) * TT + s];
        int nl = LR_map[((size_t)b * TT + t) * TT + s];
        wv += lr1[((size_t)bh * TT + t) * 64 + nl] +
              lr2[((size_t)bh * 64 + nl) * TT + s];
        wv *= 0.70710678118654752f;        // 1/sqrt(SQRT)
      }
      sc[m][s] = f2bf_u(wv);
    }
  }
  __syncthreads();

  // ---- phase 2: softmax (store unnormalized exp; fold bucket accumulation)
  {
    int row = tid >> 3, seg = tid & 7;
    int t = t0 + row;
    float mx = -3.0e38f;
    for (int s = seg; s < Lr; s += 8) mx = fmaxf(mx, bf2f_u(sc[row][s]));
    red[row][seg] = mx;
    __syncthreads();
    if (seg == 0) {
      float m2 = red[row][0];
      for (int i = 1; i < 8; ++i) m2 = fmaxf(m2, red[row][i]);
      rowmax[row] = m2;
    }
    __syncthreads();
    float rm = rowmax[row];
    float sum = 0.f;
    for (int s = seg; s < Lr; s += 8) {
      float p = __expf(bf2f_u(sc[row][s]) - rm);
      sc[row][s] = f2bf_u(p);
      sum += p;
      if (s <= t) {
        int dd = t - s;
        int rid = 32 - (dd < 32 ? dd : 32);
        atomicAdd(&bucket[row][rid], p);
      }
    }
    red[row][seg] = sum;
    __syncthreads();
    if (seg == 0) {
      float s2 = 0.f;
      for (int i = 0; i < 8; ++i) s2 += red[row][i];
      rowsum[row] = s2;
    }
  }
  __syncthreads();

  // ---- phase 3a: rel_v term via buckets: rvs = bucket @ rel_v_emb
  for (int i = tid; i < 16 * 64; i += 128) {
    int row = i >> 6, d = i & 63;
    float acc = 0.f;
    for (int r = 0; r <= 32; ++r) acc += bucket[row][r] * rel_v_emb[r * 64 + d];
    rvs[row][d] = acc;
  }
  __syncthreads();

  // ---- phase 3b: out = P @ V (each wave owns 16 d-columns; V is [B,H,D,T])
  int d0 = w * 16;
  f32x8 acc = {};
  for (int s0 = 0; s0 < Lr; s0 += 32) {
    AFrag ap, bv;
    int m = lane & 15;
    int d = d0 + (lane & 15);
    const unsigned* vrow = (const unsigned*)(Vt + ((size_t)bh * 64 + d) * TT);
#pragma unroll
    for (int p = 0; p < 8; ++p) {
      int ss = s0 + kpair(p, kbase);
      ap.u[p] = *(const unsigned*)(&sc[m][ss]);
      bv.u[p] = vrow[ss >> 1];
    }
    acc = wmma_bf16(ap, bv, acc);
  }
#pragma unroll
  for (int v = 0; v < 8; ++v) {
    int row = v + 8 * hi;
    int t = t0 + row;
    int d = d0 + (lane & 15);
    float val = (acc[v] + rvs[row][d]) / rowsum[row];
    Ob[((size_t)b * TT + t) * NXX + h * 64 + d] = f2bf_u(val);  // [B,T,H*D]
  }
}

// ---------------------------------------------------------------------------
extern "C" void kernel_launch(void* const* d_in, const int* in_sizes, int n_in,
                              void* d_out, int out_size, void* d_ws, size_t ws_size,
                              hipStream_t stream) {
  const float* x         = (const float*)d_in[0];
  const int*   rel       = (const int*)  d_in[1];
  const float* tds       = (const float*)d_in[2];
  const float* LR_Q      = (const float*)d_in[3];
  const float* LR_K      = (const float*)d_in[4];
  const int*   LR_map    = (const int*)  d_in[5];
  const float* c_attn_w  = (const float*)d_in[6];
  const float* c_attn_b  = (const float*)d_in[7];
  const float* c_proj_w  = (const float*)d_in[8];
  const float* c_proj_b  = (const float*)d_in[9];
  const float* rel_w_emb = (const float*)d_in[10];
  const float* rel_k_emb = (const float*)d_in[11];
  const float* rel_v_emb = (const float*)d_in[12];
  float* out = (float*)d_out;

  char* ws = (char*)d_ws;
  size_t off = 0;
  auto alloc = [&](size_t bytes) {
    char* p = ws + off;
    off += (bytes + 255) & ~(size_t)255;
    return p;
  };
  unsigned short* xb    = (unsigned short*)alloc((size_t)4096 * 1024 * 2);
  unsigned short* wqkvT = (unsigned short*)alloc((size_t)3072 * 1024 * 2);
  unsigned short* wprjT = (unsigned short*)alloc((size_t)1024 * 1024 * 2);
  unsigned short* Qb    = (unsigned short*)alloc((size_t)64 * 1024 * 64 * 2);
  unsigned short* Kb    = (unsigned short*)alloc((size_t)64 * 1024 * 64 * 2);
  unsigned short* Vt    = (unsigned short*)alloc((size_t)64 * 64 * 1024 * 2);
  float* qrel = (float*)alloc((size_t)64 * 1024 * QREL_N * 4);
  float* lr1  = (float*)alloc((size_t)64 * 1024 * 64 * 4);
  float* lr2  = (float*)alloc((size_t)64 * 1024 * 64 * 4);
  unsigned short* Ob = (unsigned short*)alloc((size_t)4096 * 1024 * 2);

  cvt_f32_bf16<<<(4096 * 1024 + 255) / 256, 256, 0, stream>>>(x, xb, 4096 * 1024);
  // weights transposed once: [K][N] f32 -> [N][K] bf16
  cvt_transpose_bf16<<<dim3(12, 1024), 256, 0, stream>>>(c_attn_w, wqkvT, 1024, 3072);
  cvt_transpose_bf16<<<dim3(4, 1024), 256, 0, stream>>>(c_proj_w, wprjT, 1024, 1024);

  // QKV projection: [4096,1024] x [1024,3072], 64x128 tiles, TDM double-buffered
  gemm_bf16<<<dim3(24, 64), 256, 0, stream>>>(xb, wqkvT, c_attn_b, 4096, 3072, 1024,
                                              0, nullptr, Qb, Kb, Vt);
  aux_q<<<dim3(16, 64), 128, 0, stream>>>(Qb, rel_k_emb, LR_K, qrel, lr1);
  aux_k<<<dim3(16, 64), 128, 0, stream>>>(Kb, LR_Q, lr2);

  attn_kernel<<<dim3(64, 64), 128, 0, stream>>>(Qb, Kb, Vt, rel, tds, LR_map,
                                                rel_w_emb, rel_v_emb, qrel, lr1,
                                                lr2, Ob);

  // Output projection: [4096,1024] x [1024,1024] -> fp32
  gemm_bf16<<<dim3(8, 64), 256, 0, stream>>>(Ob, wprjT, c_proj_b, 4096, 1024, 1024,
                                             1, out, nullptr, nullptr, nullptr);
}
